// QwenAttention_69303592289025
// MI455X (gfx1250) — matrix operand order repaired and checked by reading, not compiled, not run on hardware
//
#include <hip/hip_runtime.h>
#include <hip/hip_bf16.h>
#include <stdint.h>

// ---------------- CDNA5 WMMA plumbing (wave32, 16x16x32 bf16) ----------------
typedef __attribute__((ext_vector_type(16))) __bf16 v16bf;
typedef __attribute__((ext_vector_type(8)))  float  v8f;

union Frag16 { v16bf v; uint32_t u[8]; };

static __device__ __forceinline__ uint16_t f2bfu(float x) {
  union { float f; uint32_t u; } c; c.f = x;
  uint32_t r = c.u + 0x7FFFu + ((c.u >> 16) & 1u);   // round-to-nearest-even
  return (uint16_t)(r >> 16);
}
static __device__ __forceinline__ float bfu2f(uint16_t b) {
  union { uint32_t u; float f; } c; c.u = ((uint32_t)b) << 16;
  return c.f;
}
static __device__ __forceinline__ uint32_t pack2(float a, float b) {
  return (uint32_t)f2bfu(a) | ((uint32_t)f2bfu(b) << 16);
}

// A-matrix 16x32 (MxK), 16-bit: lane m=lane&15; K-halves by lane>>4.
static __device__ __forceinline__ v16bf load_a_frag(const uint16_t* tile, int row0,
                                                    int stride, int lane) {
  const int m  = lane & 15;
  const int kb = (lane >> 4) << 3;
  const uint16_t* r = tile + (row0 + m) * stride;
  Frag16 f;
#pragma unroll
  for (int i = 0; i < 4; ++i) {
    f.u[i]     = *(const uint32_t*)(r + kb + 2 * i);
    f.u[i + 4] = *(const uint32_t*)(r + 16 + kb + 2 * i);
  }
  return f.v;
}
// B-matrix 32x16 (KxN), 16-bit: lane n=lane&15; K-halves by lane>>4 (same as A).
// tile is stored [n][k] row-major (i.e. column-major of B).
static __device__ __forceinline__ v16bf load_b_frag(const uint16_t* tile, int col0,
                                                    int stride, int lane) {
  const int n  = lane & 15;
  const int kb = (lane >> 4) << 3;
  const uint16_t* r = tile + (col0 + n) * stride;
  Frag16 f;
#pragma unroll
  for (int i = 0; i < 4; ++i) {
    f.u[i]     = *(const uint32_t*)(r + kb + 2 * i);
    f.u[i + 4] = *(const uint32_t*)(r + 16 + kb + 2 * i);
  }
  return f.v;
}
static __device__ __forceinline__ v8f wmma_bf16(v16bf a, v16bf b, v8f c) {
  return __builtin_amdgcn_wmma_f32_16x16x32_bf16(false, a, false, b, (short)0, c,
                                                 false, false);
}

// async bf16 copy global -> LDS, 16B per lane (ASYNCcnt-tracked)
static __device__ __forceinline__ void async_copy_b128(uint16_t* lds_dst,
                                                       const uint16_t* gsrc) {
  const uint32_t ldsaddr = (uint32_t)(uintptr_t)lds_dst;   // addr[31:0] = LDS offset
  const uint64_t gaddr   = (uint64_t)(uintptr_t)gsrc;
  asm volatile("global_load_async_to_lds_b128 %0, %1, off"
               :: "v"(ldsaddr), "v"(gaddr)
               : "memory");
}
static __device__ __forceinline__ void wait_async_zero() {
  asm volatile("s_wait_asynccnt 0x0" ::: "memory");
}

// ---------------- problem constants ----------------
constexpr int cB = 2, cS = 2048, cE = 2048, cKV = 4, cQPG = 4, cHD = 128;
constexpr int cNH = cKV * cQPG;            // 16 q-heads
constexpr int cM  = cB * cS;               // 4096 rows of tokens
constexpr int cNQKV = cNH * cHD + 2 * cKV * cHD;  // 3072
constexpr int TS = 34;                     // padded LDS k-stride for GEMM tiles
constexpr int TB = 128 * TS;               // one 128x32 tile buffer (uint16 elems)
constexpr int SM_GEMM = 4 * TB * 2;        // bytes: A x2 + B x2 double buffered
constexpr int AS = 130;                    // padded stride for 128x128 attn tiles
constexpr int SM_ATTN = 3 * 128 * AS * 2;  // qt + kt + vt bytes

// ---------------- kernel 0: rope tables ----------------
__global__ void rope_tab_kernel(float* __restrict__ ct, float* __restrict__ st) {
  int i = blockIdx.x * 256 + threadIdx.x;  // S*64 = 131072 entries
  int s = i >> 6, f = i & 63;
  float inv = __powf(10000.0f, -(float)(2 * f) / (float)cHD);
  float ang = (float)s * inv;
  float sv, cv;
  __sincosf(ang, &sv, &cv);
  ct[i] = cv;
  st[i] = sv;
}

// ---------------- kernel 1: fused QKV projection (bias, no rope yet) ----------------
// C[M=4096, N=3072] = X[4096,2048] * W^T + b ; W rows: [0,2048)=wq, [2048,2560)=wk, rest wv
__global__ __launch_bounds__(256) void qkv_kernel(
    const float* __restrict__ x, const float* __restrict__ wq,
    const float* __restrict__ bq, const float* __restrict__ wk,
    const float* __restrict__ bk, const float* __restrict__ wv,
    const float* __restrict__ bv, uint16_t* __restrict__ qb,
    uint16_t* __restrict__ kbuf, uint16_t* __restrict__ vbuf) {
  extern __shared__ uint16_t sm[];
  uint16_t* at = sm;             // 2 * TB
  uint16_t* bt = sm + 2 * TB;    // 2 * TB
  const int t = threadIdx.x, lane = t & 31, w = t >> 5;
  const int mt = blockIdx.x, nt = blockIdx.y;
  const int gn0 = nt * 128;

  int region; const float* wbase; const float* bbase; int nrel;
  if (gn0 < cNH * cHD)                 { region = 0; wbase = wq; bbase = bq; nrel = gn0; }
  else if (gn0 < cNH * cHD + cKV * cHD){ region = 1; wbase = wk; bbase = bk; nrel = gn0 - cNH * cHD; }
  else                                 { region = 2; wbase = wv; bbase = bv; nrel = gn0 - cNH * cHD - cKV * cHD; }

  const v8f vzero = {0.f, 0.f, 0.f, 0.f, 0.f, 0.f, 0.f, 0.f};
  v8f acc[4][2];
#pragma unroll
  for (int i = 0; i < 4; ++i)
#pragma unroll
    for (int j = 0; j < 2; ++j) acc[i][j] = vzero;

  auto load_tiles = [&](int buf, int kk) {
    const int k0 = kk * 32;
    const int kq = (t & 7) * 4;
    const int r0 = t >> 3;
#pragma unroll
    for (int jj = 0; jj < 4; ++jj) {
      const int row = r0 + 32 * jj;
      float4 fa = *(const float4*)(x + (size_t)(mt * 128 + row) * cE + k0 + kq);
      uint16_t* da = at + buf * TB + row * TS + kq;
      ((uint32_t*)da)[0] = pack2(fa.x, fa.y);
      ((uint32_t*)da)[1] = pack2(fa.z, fa.w);
      float4 fb = *(const float4*)(wbase + (size_t)(nrel + row) * cE + k0 + kq);
      uint16_t* db = bt + buf * TB + row * TS + kq;
      ((uint32_t*)db)[0] = pack2(fb.x, fb.y);
      ((uint32_t*)db)[1] = pack2(fb.z, fb.w);
    }
    // prefetch the streams a few K-steps ahead into GL2
    if (k0 + 256 < cE) {
      __builtin_prefetch(x + (size_t)(mt * 128 + r0) * cE + k0 + 256, 0, 1);
      __builtin_prefetch(wbase + (size_t)(nrel + r0) * cE + k0 + 256, 0, 1);
    }
  };

  load_tiles(0, 0);
  __syncthreads();
  int buf = 0;
  const int wm = (w & 1) * 64, wn = (w >> 1) * 32;
  for (int kk = 0; kk < cE / 32; ++kk) {
    if (kk + 1 < cE / 32) load_tiles(buf ^ 1, kk + 1);
    const uint16_t* A = at + buf * TB;
    const uint16_t* Bm = bt + buf * TB;
    v16bf af[4], bf_[2];
#pragma unroll
    for (int i = 0; i < 4; ++i) af[i] = load_a_frag(A, wm + i * 16, TS, lane);
#pragma unroll
    for (int j = 0; j < 2; ++j) bf_[j] = load_b_frag(Bm, wn + j * 16, TS, lane);
#pragma unroll
    for (int i = 0; i < 4; ++i)
#pragma unroll
      for (int j = 0; j < 2; ++j) acc[i][j] = wmma_bf16(af[i], bf_[j], acc[i][j]);
    __syncthreads();
    buf ^= 1;
  }

  // epilogue: bias + bf16 store, head-major
  const int mb = (lane >> 4) << 3, nl = lane & 15;
#pragma unroll
  for (int i = 0; i < 4; ++i)
#pragma unroll
    for (int j = 0; j < 2; ++j) {
      const int n  = wn + j * 16 + nl;
      const float bias = bbase[nrel + n];
#pragma unroll
      for (int v = 0; v < 8; ++v) {
        const int m  = wm + i * 16 + mb + v;
        const int gm = mt * 128 + m;
        const int bb = gm >> 11, s = gm & (cS - 1);
        const uint16_t o = f2bfu(acc[i][j][v] + bias);
        const int rn = nrel + n;
        if (region == 0) {
          qb[(((size_t)bb * cNH + (rn >> 7)) * cS + s) * cHD + (rn & 127)] = o;
        } else if (region == 1) {
          kbuf[(((size_t)bb * cKV + (rn >> 7)) * cS + s) * cHD + (rn & 127)] = o;
        } else {
          vbuf[(((size_t)bb * cKV + (rn >> 7)) * cS + s) * cHD + (rn & 127)] = o;
        }
      }
    }
}

// ---------------- kernel 2: flash attention w/ rope-on-load ----------------
__global__ __launch_bounds__(256) void attn_kernel(
    const uint16_t* __restrict__ qb, const uint16_t* __restrict__ kbuf,
    const uint16_t* __restrict__ vbuf, const float* __restrict__ ctab,
    const float* __restrict__ stab, uint16_t* __restrict__ ob) {
  extern __shared__ uint16_t sm[];
  uint16_t* qt = sm;                  // 128 x AS
  uint16_t* kt = sm + 128 * AS;       // 128 x AS (scores K-tile, reused as P stage)
  uint16_t* vt = sm + 2 * 128 * AS;   // 128 x AS, transposed: [h][key]
  const int t = threadIdx.x, lane = t & 31, w = t >> 5;
  const int qtile = blockIdx.x;
  const int b = blockIdx.y >> 4, head = blockIdx.y & 15, kv = head >> 2;
  const uint16_t* qh = qb + ((size_t)b * cNH + head) * cS * cHD;
  const uint16_t* kh = kbuf + ((size_t)b * cKV + kv) * cS * cHD;
  const uint16_t* vh = vbuf + ((size_t)b * cKV + kv) * cS * cHD;

  // rope-apply load of a [128 rows][128 h] bf16 tile into LDS
  auto rope_load = [&](const uint16_t* src, int s0, uint16_t* dst) {
    const int sl = t & 127, half = t >> 7;
    const int sg = s0 + sl;
    const uint16_t* row = src + (size_t)sg * cHD;
    const float* cr = ctab + sg * 64;
    const float* sr = stab + sg * 64;
#pragma unroll 8
    for (int hh = 0; hh < 64; ++hh) {
      const int h = half * 64 + hh;
      const float xa = bfu2f(row[h]);
      const float xo = bfu2f(row[half ? hh : hh + 64]);
      const float o = half ? (xa * cr[hh] + xo * sr[hh])
                           : (xa * cr[hh] - xo * sr[hh]);
      dst[sl * AS + h] = f2bfu(o);
    }
  };

  const v8f vzero = {0.f, 0.f, 0.f, 0.f, 0.f, 0.f, 0.f, 0.f};
  float mrow[8], lrow[8];
  v8f o[8];
#pragma unroll
  for (int v = 0; v < 8; ++v) { mrow[v] = -3.0e38f; lrow[v] = 0.f; }
#pragma unroll
  for (int h = 0; h < 8; ++h) o[h] = vzero;

  rope_load(qh, qtile * 128, qt);
  __syncthreads();

  const int rb = w * 16;
  const int mb = (lane >> 4) << 3, nl = lane & 15;
  v16bf aq[4];
#pragma unroll
  for (int kc = 0; kc < 4; ++kc) aq[kc] = load_a_frag(qt + kc * 32, rb, AS, lane);

  const float scale = 0.088388347648318447f;  // 1/sqrt(128)

  for (int j = 0; j <= qtile; ++j) {
    rope_load(kh, j * 128, kt);
    {
      const int sl = t & 127, half = t >> 7;
      const uint16_t* row = vh + (size_t)(j * 128 + sl) * cHD + half * 64;
#pragma unroll 8
      for (int hh = 0; hh < 64; ++hh) vt[(half * 64 + hh) * AS + sl] = row[hh];
    }
    __syncthreads();

    // scores S = Q K^T (16 rows per wave x 128 keys)
    v8f sc[8];
#pragma unroll
    for (int jt = 0; jt < 8; ++jt) {
      v8f a = vzero;
#pragma unroll
      for (int kc = 0; kc < 4; ++kc)
        a = wmma_bf16(aq[kc], load_b_frag(kt + kc * 32, jt * 16, AS, lane), a);
      sc[jt] = a;
    }

    const bool diag = (j == qtile);
#pragma unroll
    for (int jt = 0; jt < 8; ++jt)
#pragma unroll
      for (int v = 0; v < 8; ++v) {
        float val = sc[jt][v] * scale;
        if (diag && (jt * 16 + nl > rb + mb + v)) val = -3.0e38f;
        sc[jt][v] = val;
      }

    // online softmax: row stats live in the lane's 16-lane half
    float mloc[8];
#pragma unroll
    for (int v = 0; v < 8; ++v) {
      float mx = sc[0][v];
#pragma unroll
      for (int jt = 1; jt < 8; ++jt) mx = fmaxf(mx, sc[jt][v]);
      mloc[v] = mx;
    }
#pragma unroll
    for (int off = 1; off < 16; off <<= 1)
#pragma unroll
      for (int v = 0; v < 8; ++v)
        mloc[v] = fmaxf(mloc[v], __shfl_xor(mloc[v], off, 32));

    float alpha[8];
#pragma unroll
    for (int v = 0; v < 8; ++v) {
      const float mnew = fmaxf(mrow[v], mloc[v]);
      alpha[v] = __expf(mrow[v] - mnew);
      mrow[v] = mnew;
    }
    float rs[8];
#pragma unroll
    for (int v = 0; v < 8; ++v) rs[v] = 0.f;
#pragma unroll
    for (int jt = 0; jt < 8; ++jt)
#pragma unroll
      for (int v = 0; v < 8; ++v) {
        const float p = __expf(sc[jt][v] - mrow[v]);
        sc[jt][v] = p;
        rs[v] += p;
      }
#pragma unroll
    for (int off = 1; off < 16; off <<= 1)
#pragma unroll
      for (int v = 0; v < 8; ++v) rs[v] += __shfl_xor(rs[v], off, 32);
#pragma unroll
    for (int v = 0; v < 8; ++v) lrow[v] = lrow[v] * alpha[v] + rs[v];
#pragma unroll
    for (int ht = 0; ht < 8; ++ht)
#pragma unroll
      for (int v = 0; v < 8; ++v) o[ht][v] *= alpha[v];

    // stage P (C-layout -> A-layout) through the now-free K tile
    __syncthreads();
#pragma unroll
    for (int jt = 0; jt < 8; ++jt)
#pragma unroll
      for (int v = 0; v < 8; ++v)
        kt[(rb + mb + v) * AS + jt * 16 + nl] = f2bfu(sc[jt][v]);

    // O += P V
    v16bf ap[4];
#pragma unroll
    for (int kc = 0; kc < 4; ++kc) ap[kc] = load_a_frag(kt + kc * 32, rb, AS, lane);
#pragma unroll
    for (int ht = 0; ht < 8; ++ht)
#pragma unroll
      for (int kc = 0; kc < 4; ++kc)
        o[ht] = wmma_bf16(ap[kc], load_b_frag(vt + kc * 32, ht * 16, AS, lane), o[ht]);

    __syncthreads();  // before next iteration overwrites kt/vt
  }

  // epilogue: normalize and store bf16 attn output [B*S, 2048]
#pragma unroll
  for (int ht = 0; ht < 8; ++ht)
#pragma unroll
    for (int v = 0; v < 8; ++v) {
      const int sg = qtile * 128 + rb + mb + v;
      const int col = head * cHD + ht * 16 + nl;
      ob[((size_t)b * cS + sg) * cE + col] = f2bfu(o[ht][v] / lrow[v]);
    }
}

// ---------------- kernel 3: output projection ----------------
// out[4096,2048] = attn(bf16) * wo^T(f32->bf16), f32 result
// A tile is a pure bf16 copy -> use ASYNCcnt-tracked global_load_async_to_lds_b128
__global__ __launch_bounds__(256) void oproj_kernel(
    const uint16_t* __restrict__ attn, const float* __restrict__ wo,
    float* __restrict__ out) {
  extern __shared__ uint16_t sm[];
  uint16_t* at = sm;
  uint16_t* bt = sm + 2 * TB;
  const int t = threadIdx.x, lane = t & 31, w = t >> 5;
  const int mt = blockIdx.x, nt = blockIdx.y;

  const v8f vzero = {0.f, 0.f, 0.f, 0.f, 0.f, 0.f, 0.f, 0.f};
  v8f acc[4][2];
#pragma unroll
  for (int i = 0; i < 4; ++i)
#pragma unroll
    for (int j = 0; j < 2; ++j) acc[i][j] = vzero;

  auto load_tiles = [&](int buf, int kk) {
    const int k0 = kk * 32;
    {  // A: already bf16 -> async DMA into LDS (no VGPR round-trip)
      const int kq = (t & 3) * 8;
      const int r0 = t >> 2;
#pragma unroll
      for (int jj = 0; jj < 2; ++jj) {
        const int row = r0 + 64 * jj;
        async_copy_b128(at + buf * TB + row * TS + kq,
                        attn + (size_t)(mt * 128 + row) * cE + k0 + kq);
      }
    }
    {  // B: wo f32 -> bf16 through VGPRs
      const int kq = (t & 7) * 4;
      const int r0 = t >> 3;
#pragma unroll
      for (int jj = 0; jj < 4; ++jj) {
        const int row = r0 + 32 * jj;
        float4 fb = *(const float4*)(wo + (size_t)(nt * 128 + row) * cE + k0 + kq);
        uint32_t* d = (uint32_t*)(bt + buf * TB + row * TS + kq);
        d[0] = pack2(fb.x, fb.y);
        d[1] = pack2(fb.z, fb.w);
      }
      if (k0 + 256 < cE)
        __builtin_prefetch(wo + (size_t)(nt * 128 + r0) * cE + k0 + 256, 0, 1);
    }
  };

  load_tiles(0, 0);
  wait_async_zero();
  __syncthreads();
  int buf = 0;
  const int wm = (w & 1) * 64, wn = (w >> 1) * 32;
  for (int kk = 0; kk < cE / 32; ++kk) {
    if (kk + 1 < cE / 32) load_tiles(buf ^ 1, kk + 1);
    const uint16_t* A = at + buf * TB;
    const uint16_t* Bm = bt + buf * TB;
    v16bf af[4], bf_[2];
#pragma unroll
    for (int i = 0; i < 4; ++i) af[i] = load_a_frag(A, wm + i * 16, TS, lane);
#pragma unroll
    for (int j = 0; j < 2; ++j) bf_[j] = load_b_frag(Bm, wn + j * 16, TS, lane);
#pragma unroll
    for (int i = 0; i < 4; ++i)
#pragma unroll
      for (int j = 0; j < 2; ++j) acc[i][j] = wmma_bf16(af[i], bf_[j], acc[i][j]);
    wait_async_zero();   // next-buffer async LDS writes landed
    __syncthreads();
    buf ^= 1;
  }

  const int mb = (lane >> 4) << 3, nl = lane & 15;
#pragma unroll
  for (int i = 0; i < 4; ++i)
#pragma unroll
    for (int j = 0; j < 2; ++j) {
      const int gn = nt * 128 + wn + j * 16 + nl;
#pragma unroll
      for (int v = 0; v < 8; ++v) {
        const int gm = mt * 128 + wm + i * 16 + mb + v;
        out[(size_t)gm * cE + gn] = acc[i][j][v];
      }
    }
}

// ---------------- host launcher ----------------
extern "C" void kernel_launch(void* const* d_in, const int* in_sizes, int n_in,
                              void* d_out, int out_size, void* d_ws, size_t ws_size,
                              hipStream_t stream) {
  const float* x  = (const float*)d_in[0];
  const float* wq = (const float*)d_in[1];
  const float* bq = (const float*)d_in[2];
  const float* wk = (const float*)d_in[3];
  const float* bk = (const float*)d_in[4];
  const float* wv = (const float*)d_in[5];
  const float* bv = (const float*)d_in[6];
  const float* wo = (const float*)d_in[7];
  float* out = (float*)d_out;

  float* ctab = (float*)d_ws;                       // S*64 f32
  float* stab = ctab + cS * 64;                     // S*64 f32
  uint16_t* qbuf = (uint16_t*)(stab + cS * 64);     // B*16*S*128 bf16
  uint16_t* kbuf = qbuf + (size_t)cB * cNH * cS * cHD;
  uint16_t* vbuf = kbuf + (size_t)cB * cKV * cS * cHD;
  uint16_t* attn = vbuf + (size_t)cB * cKV * cS * cHD;  // [B*S, 2048] bf16

  (void)hipFuncSetAttribute((const void*)attn_kernel,
                            hipFuncAttributeMaxDynamicSharedMemorySize, SM_ATTN);

  rope_tab_kernel<<<(cS * 64) / 256, 256, 0, stream>>>(ctab, stab);
  qkv_kernel<<<dim3(cM / 128, cNQKV / 128), 256, SM_GEMM, stream>>>(
      x, wq, bq, wk, bk, wv, bv, qbuf, kbuf, vbuf);
  attn_kernel<<<dim3(cS / 128, cB * cNH), 256, SM_ATTN, stream>>>(
      qbuf, kbuf, vbuf, ctab, stab, attn);
  oproj_kernel<<<dim3(cM / 128, cE / 128), 256, SM_GEMM, stream>>>(attn, wo, out);
}
